// Net_LIF_48180943127012
// MI455X (gfx1250) — matrix operand order repaired and checked by reading, not compiled, not run on hardware
//
#include <hip/hip_runtime.h>

// ---------------------------------------------------------------------------
// SNN 2-layer LIF forward, MI455X (gfx1250, wave32, WMMA).
// Bandwidth-bound (~750MB traffic @ 23.3TB/s): single pass over time with
// persistent membrane state in VGPRs, batch partitioned across workgroups,
// layer1->layer2 handoff via LDS, matmuls on v_wmma_f32_16x16x32_f16.
// ---------------------------------------------------------------------------

typedef __attribute__((ext_vector_type(16))) _Float16 v16h;
typedef __attribute__((ext_vector_type(8)))  _Float16 v8h;
typedef __attribute__((ext_vector_type(8)))  float    v8f;
typedef __attribute__((ext_vector_type(4)))  float    v4f;

#define T_STEPS 100
#define BATCH   1024
#define NI      256
#define NH      512
#define NO      10
#define NOPAD   16
#define BT      16    // batch rows per workgroup (one WMMA M tile)
#define WAVES   8     // 256 threads
#define NTW     4     // 16-wide N tiles per wave: 8 waves * 4 * 16 = NH

#define BETA1 0.9f
#define BETA2 0.8f
#define THR   1.0f

// ---- weight preprocessing: f32 -> f16 (values are exactly representable) ----
__global__ void prep_w1(const float* __restrict__ W1, _Float16* __restrict__ w1h) {
    int i = blockIdx.x * 256 + threadIdx.x;
    if (i < NH * NI) w1h[i] = (_Float16)W1[i];
}

// w2p[n*NH + k] = (n < NO) ? W2[n*NH + k] : 0   (pad NO=10 -> 16 columns)
__global__ void prep_w2(const float* __restrict__ W2, _Float16* __restrict__ w2p) {
    int i = blockIdx.x * 256 + threadIdx.x;      // over NOPAD*NH
    if (i < NOPAD * NH) {
        int n = i / NH;
        w2p[i] = (n < NO) ? (_Float16)W2[i] : (_Float16)0.0f;
    }
}

// ---------------------------------------------------------------------------
__global__ __launch_bounds__(256) void snn_lif_fwd(
    const float*    __restrict__ x,     // [T, B, NI] f32
    const _Float16* __restrict__ w1h,   // [NH, NI]  f16
    const _Float16* __restrict__ w2p,   // [16, NH]  f16 (padded W2)
    float*          __restrict__ out)   // cur1|cur2|spk1|spk2|mem1|mem2
{
    __shared__ _Float16 xh[BT * NI];    // x_t tile, f16   (8 KB)
    __shared__ _Float16 spkh[BT * NH];  // spk1 tile, f16  (16 KB)

    const int tid   = threadIdx.x;
    const int lane  = tid & 31;
    const int wave  = tid >> 5;
    const int bTile = blockIdx.x * BT;

    const int mrow = lane & 15;   // N index within 16-wide tile (B/C/D layout)
    const int khi  = lane >> 4;   // K-half / M-half selector

    // output section offsets (floats), return order: cur1,cur2,spk1,spk2,mem1,mem2
    const size_t S1 = (size_t)T_STEPS * BATCH * NH;
    const size_t S2 = (size_t)T_STEPS * BATCH * NO;
    float* cur1_o = out;
    float* cur2_o = out + S1;
    float* spk1_o = out + S1 + S2;
    float* spk2_o = out + 2 * S1 + S2;
    float* mem1_o = out + 2 * S1 + 2 * S2;
    float* mem2_o = out + 3 * S1 + 2 * S2;

    // persistent recurrent state (zero-init, matches reference)
    v8f mem1[NTW];
    #pragma unroll
    for (int i = 0; i < NTW; ++i) mem1[i] = (v8f){};
    v8f mem2 = (v8f){};   // live only in wave 0

    #pragma unroll 1
    for (int t = 0; t < T_STEPS; ++t) {
        // ---- stage x[t, bTile:bTile+16, :] into LDS as f16 ----
        {
            const int r  = tid >> 4;          // 0..15 batch row in tile
            const int c0 = (tid & 15) * 16;   // 16 columns per thread
            const float* src = x + ((size_t)t * BATCH + bTile + r) * NI + c0;
            _Float16* dst = xh + r * NI + c0;
            #pragma unroll
            for (int j = 0; j < 16; j += 4) {
                v4f v = *(const v4f*)(src + j);
                dst[j + 0] = (_Float16)v.x;
                dst[j + 1] = (_Float16)v.y;
                dst[j + 2] = (_Float16)v.z;
                dst[j + 3] = (_Float16)v.w;
            }
            if (t + 1 < T_STEPS)  // pull next timestep slice toward L2/L0
                __builtin_prefetch(src + (size_t)BATCH * NI, 0, 0);
        }
        __syncthreads();

        // ---- layer 1: cur1 = x_t @ W1^T over this wave's 64-wide N strip ----
        v8f c[NTW];
        #pragma unroll
        for (int i = 0; i < NTW; ++i) c[i] = (v8f){};

        #pragma unroll
        for (int kk = 0; kk < NI / 32; ++kk) {
            const int kb = kk * 32;
            // A fragment (16x32 f16): lane khi==0 -> K kb+0..15
            //                         lane khi==1 -> K kb+16..31
            union { v16h v; v8h h[2]; } au;
            const _Float16* ap = xh + mrow * NI + kb + khi * 16;
            au.h[0] = *(const v8h*)(ap);
            au.h[1] = *(const v8h*)(ap + 8);
            const v16h a = au.v;

            #pragma unroll
            for (int nt = 0; nt < NTW; ++nt) {
                const int n = (wave * NTW + nt) * 16 + mrow;   // hidden unit
                // B fragment (32x16 f16): lane holds K kb+khi*16 .. +15, contiguous
                const v16h b = *(const v16h*)(w1h + (size_t)n * NI + kb + khi * 16);
                c[nt] = __builtin_amdgcn_wmma_f32_16x16x32_f16(
                            false, a, false, b, (short)0, c[nt], false, false);
            }
        }

        // ---- LIF layer 1 + emit cur1/spk1/mem1, spk1 -> LDS ----
        #pragma unroll
        for (int nt = 0; nt < NTW; ++nt) {
            const int n = (wave * NTW + nt) * 16 + mrow;
            v8f m = mem1[nt];
            const size_t rowBase = ((size_t)t * BATCH + bTile) * NH;
            #pragma unroll
            for (int e = 0; e < 8; ++e) {
                const float reset = (m[e] > THR) ? THR : 0.0f;   // from prev mem
                const float mn    = BETA1 * m[e] + c[nt][e] - reset;
                m[e] = mn;
                const float spk = (mn > THR) ? 1.0f : 0.0f;
                const int   br  = e + khi * 8;                    // batch row in tile
                const size_t idx = rowBase + (size_t)br * NH + n;
                cur1_o[idx] = c[nt][e];
                spk1_o[idx] = spk;
                mem1_o[idx] = mn;
                spkh[br * NH + n] = (_Float16)spk;
            }
            mem1[nt] = m;
        }
        __syncthreads();

        // ---- layer 2 on wave 0: cur2 = spk1 @ W2^T (NO padded to 16) ----
        if (wave == 0) {
            v8f c2 = (v8f){};
            #pragma unroll
            for (int kk = 0; kk < NH / 32; ++kk) {
                const int kb = kk * 32;
                union { v16h v; v8h h[2]; } au;
                const _Float16* ap = spkh + mrow * NH + kb + khi * 16;
                au.h[0] = *(const v8h*)(ap);
                au.h[1] = *(const v8h*)(ap + 8);
                const v16h b2 = *(const v16h*)(w2p + (size_t)mrow * NH + kb + khi * 16);
                c2 = __builtin_amdgcn_wmma_f32_16x16x32_f16(
                         false, au.v, false, b2, (short)0, c2, false, false);
            }
            v8f m2 = mem2;
            v8f s2;
            #pragma unroll
            for (int e = 0; e < 8; ++e) {
                const float reset = (m2[e] > THR) ? THR : 0.0f;
                const float mn    = BETA2 * m2[e] + c2[e] - reset;
                m2[e] = mn;
                s2[e] = (mn > THR) ? 1.0f : 0.0f;
            }
            mem2 = m2;
            if (mrow < NO) {   // after all WMMAs; only real output columns
                const size_t rb = ((size_t)t * BATCH + bTile) * NO;
                #pragma unroll
                for (int e = 0; e < 8; ++e) {
                    const size_t idx = rb + (size_t)(e + khi * 8) * NO + mrow;
                    cur2_o[idx] = c2[e];
                    spk2_o[idx] = s2[e];
                    mem2_o[idx] = m2[e];
                }
            }
        }
        __syncthreads();   // protect xh/spkh before next iteration rewrites them
    }
}

// ---------------------------------------------------------------------------
extern "C" void kernel_launch(void* const* d_in, const int* in_sizes, int n_in,
                              void* d_out, int out_size, void* d_ws, size_t ws_size,
                              hipStream_t stream) {
    (void)in_sizes; (void)n_in; (void)out_size; (void)ws_size;
    const float* x  = (const float*)d_in[0];   // [T,B,NI]
    const float* W1 = (const float*)d_in[1];   // [NH,NI]
    const float* W2 = (const float*)d_in[2];   // [NO,NH]
    float* out = (float*)d_out;

    _Float16* w1h = (_Float16*)d_ws;           // NH*NI f16 (256 KB)
    _Float16* w2p = w1h + (size_t)NH * NI;     // NOPAD*NH f16 (16 KB)

    prep_w1<<<(NH * NI + 255) / 256, 256, 0, stream>>>(W1, w1h);
    prep_w2<<<(NOPAD * NH + 255) / 256, 256, 0, stream>>>(W2, w2p);
    snn_lif_fwd<<<BATCH / BT, 256, 0, stream>>>(x, w1h, w2p, out);
}
